// CDANN_76854144795197
// MI455X (gfx1250) — compile-verified, matrix operand order checked by reading, not run on hardware
//
#include <hip/hip_runtime.h>

typedef __attribute__((ext_vector_type(16))) _Float16 v16h;
typedef __attribute__((ext_vector_type(2)))  _Float16 v2h;
typedef __attribute__((ext_vector_type(8)))  float    v8f;

#define WPB   8     // waves per block
#define NFRAG 36    // total B fragments

// K index inside a 16x32 f16 A-fragment / 32x16 f16 B-fragment (CDNA5 ISA 7.12.2)
__device__ __forceinline__ int kslot(int j, int group) {
  int v = j >> 1, hi = j & 1;
  return ((v & 4) << 2) + (group << 3) + ((v & 3) << 1) + hi;
}

// half-offset (lane-invariant part) within an A-layout staged fragment for column k:
// full half index = kt*512 + m*16 + aslot_off(k&31)
__device__ __forceinline__ int aslot_off(int kp) {
  int gp = (kp >> 3) & 1;
  int v  = ((kp & 16) >> 2) | ((kp >> 1) & 3);
  int j  = v * 2 + (kp & 1);
  return gp * 256 + j;
}

// packed f16 ReLU in one VALU op; inline asm so the compiler cannot
// re-expand it into canonicalize + max
__device__ __forceinline__ v2h relu2(v2h p) {
  v2h r;
  asm("v_pk_max_num_f16 %0, %1, 0" : "=v"(r) : "v"(p));
  return r;
}

// Build B fragment (32x16 f16) of W[Krows x Ncols] fp32 row-major, tile (kbase,nbase).
// Row k==Krows carries bias[n] (bias folding via ones-column in A); rows beyond are 0,
// so stale/padded A-slots contribute exactly 0.
__device__ __forceinline__ v16h bfrag(const float* __restrict__ W,
                                      const float* __restrict__ bias,
                                      int Krows, int Ncols,
                                      int kbase, int nbase, int lane) {
  v16h b = {};
  int n = nbase + (lane & 15);
  int g = (lane >> 4) & 1;
  if (n < Ncols) {
#pragma unroll
    for (int j = 0; j < 16; ++j) {
      int k = kbase + kslot(j, g);
      float w = 0.f;
      if (k < Krows) w = W[k * Ncols + n];
      else if (bias && k == Krows) w = bias[n];
      b[j] = (_Float16)w;
    }
  }
  return b;
}

__device__ __forceinline__ v8f wmma16(v16h a, v16h b, v8f c) {
  return __builtin_amdgcn_wmma_f32_16x16x32_f16(false, a, false, b, (short)0, c,
                                                false, false);
}

// D (f32 16x16: lane->N, vgpr r -> M=r+8*(lane>>4)) -> f16 ReLU ->
// next consumer's A-fragment layout in LDS. Bias already folded into the WMMA.
__device__ __forceinline__ void store_act(_Float16* fragbase, const v8f& d,
                                          int nbase, int Ncols, int lane) {
  int n = nbase + (lane & 15);
  int g = lane >> 4;
  if (n < Ncols) {
    _Float16* base = fragbase + (n >> 5) * 512 + aslot_off(n & 31) + g * 128;
#pragma unroll
    for (int i = 0; i < 4; ++i) {
      v2h p;
      p[0] = (_Float16)d[2 * i];
      p[1] = (_Float16)d[2 * i + 1];
      p = relu2(p);
      base[(2 * i) * 16]     = p[0];     // ds_store_b16
      base[(2 * i + 1) * 16] = p[1];     // ds_store_b16_d16_hi
    }
  }
}

// f2 variant: explicit bias (K=64, no padding slack), then f16 ReLU
__device__ __forceinline__ void store_act_bias(_Float16* fragbase, const v8f& d,
                                               const float* __restrict__ bias,
                                               int nbase, int Ncols, int lane) {
  int n = nbase + (lane & 15);
  int g = lane >> 4;
  if (n < Ncols) {
    float bv = bias[n];
    _Float16* base = fragbase + (n >> 5) * 512 + aslot_off(n & 31) + g * 128;
#pragma unroll
    for (int i = 0; i < 4; ++i) {
      v2h p;
      p[0] = (_Float16)(d[2 * i] + bv);
      p[1] = (_Float16)(d[2 * i + 1] + bv);
      p = relu2(p);
      base[(2 * i) * 16]     = p[0];
      base[(2 * i + 1) * 16] = p[1];
    }
  }
}

// D -> per-wave f32 staging buffer (row-major [16][Ncols]); bias already folded
__device__ __forceinline__ void stage_out(float* so, const v8f& d, int Ncols, int lane) {
  int n = lane & 15, g = lane >> 4;
  if (n < Ncols) {
#pragma unroll
    for (int r = 0; r < 8; ++r) so[(r + 8 * g) * Ncols + n] = d[r];
  }
}

// Compiler scheduling fence only: same-wave DS ops execute in order in HW
// (ISA 7.3), so no s_wait_dscnt is needed between store and dependent load.
__device__ __forceinline__ void lds_fence() {
  __builtin_amdgcn_wave_barrier();
}

// Fragment index map:
//  0-3  Bf1[nt]            4-9  Bf2[kt*3+nt]      10-13 Bt1[kt*2+nt]   14 Bt2
// 15-18 Bp1[kt*2+nt]      19-20 Bp2[nt]           21 Bp3
// 22+7e+kt*2+nt Bc1[e]    26+7e+nt Bc2[e]         28+7e Bc3[e]
__device__ v16h build_frag(int f, int lane,
                           const float* Wf1, const float* bf1,
                           const float* Wf2,
                           const float* Wt1, const float* bt1,
                           const float* Wt2, const float* bt2,
                           const float* Wp1, const float* bp1,
                           const float* Wp2, const float* bp2,
                           const float* Wp3, const float* bp3,
                           const float* Wc1, const float* bc1,
                           const float* Wc2, const float* bc2,
                           const float* Wc3, const float* bc3) {
  switch (f) {
    case 0:  return bfrag(Wf1, bf1, 30, 64, 0, 0, lane);
    case 1:  return bfrag(Wf1, bf1, 30, 64, 0, 16, lane);
    case 2:  return bfrag(Wf1, bf1, 30, 64, 0, 32, lane);
    case 3:  return bfrag(Wf1, bf1, 30, 64, 0, 48, lane);
    case 4:  return bfrag(Wf2, nullptr, 64, 42, 0, 0, lane);
    case 5:  return bfrag(Wf2, nullptr, 64, 42, 0, 16, lane);
    case 6:  return bfrag(Wf2, nullptr, 64, 42, 0, 32, lane);
    case 7:  return bfrag(Wf2, nullptr, 64, 42, 32, 0, lane);
    case 8:  return bfrag(Wf2, nullptr, 64, 42, 32, 16, lane);
    case 9:  return bfrag(Wf2, nullptr, 64, 42, 32, 32, lane);
    case 10: return bfrag(Wt1, bt1, 42, 20, 0, 0, lane);
    case 11: return bfrag(Wt1, bt1, 42, 20, 0, 16, lane);
    case 12: return bfrag(Wt1, bt1, 42, 20, 32, 0, lane);   // k=42 row = bt1
    case 13: return bfrag(Wt1, bt1, 42, 20, 32, 16, lane);
    case 14: return bfrag(Wt2, bt2, 20, 2, 0, 0, lane);     // k=20 row = bt2
    case 15: return bfrag(Wp1, bp1, 42, 20, 0, 0, lane);
    case 16: return bfrag(Wp1, bp1, 42, 20, 0, 16, lane);
    case 17: return bfrag(Wp1, bp1, 42, 20, 32, 0, lane);
    case 18: return bfrag(Wp1, bp1, 42, 20, 32, 16, lane);
    case 19: return bfrag(Wp2, bp2, 20, 20, 0, 0, lane);
    case 20: return bfrag(Wp2, bp2, 20, 20, 0, 16, lane);
    case 21: return bfrag(Wp3, bp3, 20, 6, 0, 0, lane);
    case 22: return bfrag(Wc1, bc1, 42, 20, 0, 0, lane);
    case 23: return bfrag(Wc1, bc1, 42, 20, 0, 16, lane);
    case 24: return bfrag(Wc1, bc1, 42, 20, 32, 0, lane);
    case 25: return bfrag(Wc1, bc1, 42, 20, 32, 16, lane);
    case 26: return bfrag(Wc2, bc2, 20, 20, 0, 0, lane);
    case 27: return bfrag(Wc2, bc2, 20, 20, 0, 16, lane);
    case 28: return bfrag(Wc3, bc3, 20, 6, 0, 0, lane);
    case 29: return bfrag(Wc1 + 840, bc1 + 20, 42, 20, 0, 0, lane);
    case 30: return bfrag(Wc1 + 840, bc1 + 20, 42, 20, 0, 16, lane);
    case 31: return bfrag(Wc1 + 840, bc1 + 20, 42, 20, 32, 0, lane);
    case 32: return bfrag(Wc1 + 840, bc1 + 20, 42, 20, 32, 16, lane);
    case 33: return bfrag(Wc2 + 400, bc2 + 20, 20, 20, 0, 0, lane);
    case 34: return bfrag(Wc2 + 400, bc2 + 20, 20, 20, 0, 16, lane);
    default: return bfrag(Wc3 + 120, bc3 + 6, 20, 6, 0, 0, lane);
  }
}

__global__ void __launch_bounds__(256, 4)
cdann_fused_wmma(const float* __restrict__ x, const int* __restrict__ y,
                 const float* __restrict__ Wf1, const float* __restrict__ bf1,
                 const float* __restrict__ Wf2, const float* __restrict__ bf2,
                 const float* __restrict__ Wt1, const float* __restrict__ bt1,
                 const float* __restrict__ Wt2, const float* __restrict__ bt2,
                 const float* __restrict__ Wp1, const float* __restrict__ bp1,
                 const float* __restrict__ Wp2, const float* __restrict__ bp2,
                 const float* __restrict__ Wp3, const float* __restrict__ bp3,
                 const float* __restrict__ Wc1, const float* __restrict__ bc1,
                 const float* __restrict__ Wc2, const float* __restrict__ bc2,
                 const float* __restrict__ Wc3, const float* __restrict__ bc3,
                 float* __restrict__ out_task, float* __restrict__ out_pd,
                 float* __restrict__ out_cd, int ntiles) {
  __shared__ v16h s_bf[NFRAG * 32];                       // shared weight fragments
  // per-wave: frags 0-1 = x / h / fea ; frag 2 = all 20-wide hidden stages
  __shared__ __align__(32) _Float16 s_act[WPB][1536];
  __shared__ float s_out[WPB][96];                        // per-wave output staging

  const int lane = threadIdx.x & 31;
  const int wave = threadIdx.x >> 5;
  const int g    = lane >> 4;
  const int n0   = lane & 15;
  _Float16* act  = s_act[wave];
  _Float16* hid  = act + 1024;                            // frag 2
  float*    so   = s_out[wave];
  const _Float16 one  = (_Float16)1.f;
  const _Float16 zero = (_Float16)0.f;

#define LDB(f) (s_bf[(f) * 32 + lane])
#define LDA(i) (((const v16h*)act)[(i) * 32 + lane])

  // build shared B fragments once per block
  for (int f = wave; f < NFRAG; f += WPB)
    s_bf[f * 32 + lane] = build_frag(f, lane, Wf1, bf1, Wf2, Wt1, bt1, Wt2, bt2,
                                     Wp1, bp1, Wp2, bp2, Wp3, bp3,
                                     Wc1, bc1, Wc2, bc2, Wc3, bc3);
  // zero act buffer so padded/stale slots are finite from tile 0 on
  {
    v16h z = {};
    ((v16h*)act)[lane] = z;
    ((v16h*)act)[32 + lane] = z;
    ((v16h*)act)[64 + lane] = z;
  }
  // frag2 ones col @ k=20 (bias row for all 20-wide stages); slots 21..31 stay 0.
  // Nothing ever writes frag2 k'>=20 again, so this lives for the whole kernel.
  if (lane < 16) hid[lane * 16 + aslot_off(20)] = one;
  __syncthreads();

  const int gwave  = blockIdx.x * WPB + wave;
  const int nwaves = gridDim.x * WPB;

  for (int t = gwave; t < ntiles; t += nwaves) {
    const int row0 = t * 16;

    int tn = t + nwaves;                    // prefetch next x tile
    if (tn < ntiles) __builtin_prefetch(x + (size_t)tn * 480 + lane * 16, 0, 0);

    // ---- coalesced x-tile load, scattered f16 store into A-frag layout ----
    const float* xs = x + (size_t)row0 * 30;
#pragma unroll
    for (int i = 0; i < 15; ++i) {
      int idx = lane + i * 32;              // 0..479 contiguous
      int m = idx / 30, k = idx - m * 30;   // k < 30
      act[m * 16 + aslot_off(k)] = (_Float16)xs[idx];
    }
    // unconditional ones column: lanes 0-15 write 1.0 @k=30 (bf1 row),
    // lanes 16-31 write 0.0 @k=31 (dead pad) -- no exec-mask churn
    act[n0 * 16 + aslot_off(30 + g)] = g ? zero : one;
    lds_fence();

    // ---- f1: [16,31]x[31,64]  (bias folded at k=30) ----
    {
      v16h A0 = LDA(0);
      __builtin_amdgcn_wave_barrier();
#pragma unroll
      for (int nt = 0; nt < 4; ++nt) {
        v8f c = {};
        c = wmma16(A0, LDB(nt), c);
        store_act(act, c, nt * 16, 64, lane);
      }
    }
    lds_fence();

    // ---- f2: [16,64]x[64,42]  (explicit bias, no K slack) ----
    {
      v16h Ah0 = LDA(0);
      v16h Ah1 = LDA(1);
      __builtin_amdgcn_wave_barrier();
#pragma unroll
      for (int nt = 0; nt < 3; ++nt) {
        v8f c = {};
        c = wmma16(Ah0, LDB(4 + nt), c);
        c = wmma16(Ah1, LDB(7 + nt), c);
        store_act_bias(act, c, bf2, nt * 16, 42, lane);
      }
      // fea ones @k=42 (lanes 0-15), zero @k=43 dead pad (lanes 16-31)
      act[512 + n0 * 16 + aslot_off(10 + g)] = g ? zero : one;
    }
    lds_fence();
    // fea now persists in frags 0-1 for the whole tile (heads write only frag 2)

    // ---- task head 42->20->2  (biases folded) ----
    {
      v16h Af0 = LDA(0);
      v16h Af1 = LDA(1);
      __builtin_amdgcn_wave_barrier();
#pragma unroll
      for (int nt = 0; nt < 2; ++nt) {
        v8f c = {};
        c = wmma16(Af0, LDB(10 + nt), c);
        c = wmma16(Af1, LDB(12 + nt), c);
        store_act(hid, c, nt * 16, 20, lane);
      }
    }
    lds_fence();
    {
      v16h At = LDA(2);
      __builtin_amdgcn_wave_barrier();
      v8f c = {};
      c = wmma16(At, LDB(14), c);
      stage_out(so, c, 2, lane);
      lds_fence();
      out_task[(size_t)row0 * 2 + lane] = so[lane];   // 128B coalesced
    }

    // ---- prior domain head 42->20->20->6  (biases folded) ----
    {
      v16h Af0 = LDA(0);
      v16h Af1 = LDA(1);
      __builtin_amdgcn_wave_barrier();
#pragma unroll
      for (int nt = 0; nt < 2; ++nt) {
        v8f c = {};
        c = wmma16(Af0, LDB(15 + nt), c);
        c = wmma16(Af1, LDB(17 + nt), c);
        store_act(hid, c, nt * 16, 20, lane);
      }
    }
    lds_fence();
    {
      v16h Ap = LDA(2);
      __builtin_amdgcn_wave_barrier();
#pragma unroll
      for (int nt = 0; nt < 2; ++nt) {
        v8f c = {};
        c = wmma16(Ap, LDB(19 + nt), c);
        store_act(hid, c, nt * 16, 20, lane);
      }
    }
    lds_fence();
    {
      v16h Ap = LDA(2);
      __builtin_amdgcn_wave_barrier();
      v8f c = {};
      c = wmma16(Ap, LDB(21), c);
      stage_out(so, c, 6, lane);
      lds_fence();
#pragma unroll
      for (int i = 0; i < 3; ++i)           // 384B coalesced
        out_pd[(size_t)row0 * 6 + i * 32 + lane] = so[i * 32 + lane];
    }

    // ---- conditional experts, routed by label via ballot mask ----
    // one coalesced y load; bit l of ymask = (y[row0+l&15] != 0)
    int yl = y[row0 + n0];
    unsigned ymask = __builtin_amdgcn_ballot_w32(yl != 0);
    unsigned my = (ymask >> (8 * g)) & 0xffu;   // bits for rows r+8g, r=0..7
#pragma unroll
    for (int e = 0; e < 2; ++e) {
      const int fb = 22 + e * 7;
      {
        v16h Af0 = LDA(0);
        v16h Af1 = LDA(1);
        __builtin_amdgcn_wave_barrier();
#pragma unroll
        for (int nt = 0; nt < 2; ++nt) {
          v8f c = {};
          c = wmma16(Af0, LDB(fb + nt), c);
          c = wmma16(Af1, LDB(fb + 2 + nt), c);
          store_act(hid, c, nt * 16, 20, lane);
        }
      }
      lds_fence();
      {
        v16h Ac = LDA(2);
        __builtin_amdgcn_wave_barrier();
#pragma unroll
        for (int nt = 0; nt < 2; ++nt) {
          v8f c = {};
          c = wmma16(Ac, LDB(fb + 4 + nt), c);
          store_act(hid, c, nt * 16, 20, lane);
        }
      }
      lds_fence();
      {
        v16h Ac = LDA(2);
        __builtin_amdgcn_wave_barrier();
        v8f c = {};
        c = wmma16(Ac, LDB(fb + 6), c);
        // e==0: unconditional; e==1: overwrite rows with y==1
        if (n0 < 6) {
#pragma unroll
          for (int r = 0; r < 8; ++r) {
            if (e == 0) so[(r + 8 * g) * 6 + n0] = c[r];
            else if ((my >> r) & 1) so[(r + 8 * g) * 6 + n0] = c[r];
          }
        }
      }
      lds_fence();
    }
#pragma unroll
    for (int i = 0; i < 3; ++i)             // 384B coalesced
      out_cd[(size_t)row0 * 6 + i * 32 + lane] = so[i * 32 + lane];
  }
#undef LDB
#undef LDA
}

extern "C" void kernel_launch(void* const* d_in, const int* in_sizes, int n_in,
                              void* d_out, int out_size, void* d_ws, size_t ws_size,
                              hipStream_t stream) {
  const float* x   = (const float*)d_in[0];
  const int*   y   = (const int*)d_in[1];
  const float* Wf1 = (const float*)d_in[2];
  const float* bf1 = (const float*)d_in[3];
  const float* Wf2 = (const float*)d_in[4];
  const float* bf2 = (const float*)d_in[5];
  const float* Wt1 = (const float*)d_in[6];
  const float* bt1 = (const float*)d_in[7];
  const float* Wt2 = (const float*)d_in[8];
  const float* bt2 = (const float*)d_in[9];
  const float* Wp1 = (const float*)d_in[10];
  const float* bp1 = (const float*)d_in[11];
  const float* Wp2 = (const float*)d_in[12];
  const float* bp2 = (const float*)d_in[13];
  const float* Wp3 = (const float*)d_in[14];
  const float* bp3 = (const float*)d_in[15];
  const float* Wc1 = (const float*)d_in[16];
  const float* bc1 = (const float*)d_in[17];
  const float* Wc2 = (const float*)d_in[18];
  const float* bc2 = (const float*)d_in[19];
  const float* Wc3 = (const float*)d_in[20];
  const float* bc3 = (const float*)d_in[21];

  const int B = in_sizes[0] / 30;           // 524288
  float* out      = (float*)d_out;
  float* out_task = out;                    // [B,2]
  float* out_pd   = out + (size_t)B * 2;    // [B,6]
  float* out_cd   = out + (size_t)B * 8;    // [B,6]

  const int ntiles = B / 16;                // 32768
  dim3 block(256);
  dim3 grid(1024);                          // 8192 waves, 4 tiles each

  cdann_fused_wmma<<<grid, block, 0, stream>>>(
      x, y, Wf1, bf1, Wf2, bf2, Wt1, bt1, Wt2, bt2,
      Wp1, bp1, Wp2, bp2, Wp3, bp3, Wc1, bc1, Wc2, bc2, Wc3, bc3,
      out_task, out_pd, out_cd, ntiles);
}
